// RefineDetMultiBoxLoss_5712306504444
// MI455X (gfx1250) — compile-verified, matrix operand order checked by reading, not run on hardware
//
#include <hip/hip_runtime.h>
#include <hip/hip_bf16.h>
#include <math.h>

#define B_    64
#define P_    16320
#define M_    24
#define C_    21
#define THRESH 0.5f
#define THETA  0.01f
#define BLK   256
#define PBLK  64   // blocks per batch: 64*256 = 16384 >= 16320

typedef __attribute__((ext_vector_type(16))) _Float16 v16h;
typedef __attribute__((ext_vector_type(8)))  float    v8f;

// ---- optional CDNA5 async global->LDS path (guarded; sync copy fallback) ----
#if defined(__has_builtin)
#  if __has_builtin(__builtin_amdgcn_global_load_async_to_lds_b32)
#    define USE_ASYNC_LDS 1
#  endif
#  if __has_builtin(__builtin_amdgcn_s_wait_asynccnt)
#    define HAVE_WAIT_ASYNC_BUILTIN 1
#  endif
#endif
#ifndef USE_ASYNC_LDS
#  define USE_ASYNC_LDS 0
#endif

// builtin signature (from hipcc diagnostic): (__device__ int*, __shared__ int*, imm, imm)
typedef __attribute__((address_space(1))) int* gas1_i32p;
typedef __attribute__((address_space(3))) int* las3_i32p;

static __device__ __forceinline__ void wait_async0() {
#if USE_ASYNC_LDS
#  if defined(HAVE_WAIT_ASYNC_BUILTIN)
    __builtin_amdgcn_s_wait_asynccnt(0);
#  else
    asm volatile("s_wait_asynccnt 0" ::: "memory");
#  endif
#endif
}

// ---------------------------------------------------------------- init
__global__ void k_init(unsigned long long* packed, int* numpos, float* sumposce,
                       float* lossl, float* lossc) {
    int t = blockIdx.x * blockDim.x + threadIdx.x;
    if (t < B_ * M_) packed[t] = 0ull;
    if (t < B_) { numpos[t] = 0; sumposce[t] = 0.f; }
    if (t == 0) { lossl[0] = 0.f; lossc[0] = 0.f; }
}

// ---------------------------------------------------------------- match
__global__ void k_match(const float4* __restrict__ arm_loc,
                        const float4* __restrict__ priors,
                        const float4* __restrict__ truths,   // [B*M]
                        float* __restrict__ bestov,
                        int*   __restrict__ bestidx,
                        unsigned long long* __restrict__ packed) {
    __shared__ float4 st[M_];
    __shared__ unsigned long long sbm[M_];
    const int b = blockIdx.y;
    const int tid = threadIdx.x;
    const int p = blockIdx.x * BLK + tid;
    if (tid < M_) { st[tid] = truths[b * M_ + tid]; sbm[tid] = 0ull; }
    __syncthreads();

    if (p < P_) {
        float4 pr = priors[p];
        float4 l  = arm_loc[(size_t)b * P_ + p];
        float cx = pr.x + l.x * 0.1f * pr.z;
        float cy = pr.y + l.y * 0.1f * pr.w;
        float w  = pr.z * expf(l.z * 0.2f);
        float h  = pr.w * expf(l.w * 0.2f);
        float dx0 = cx - w * 0.5f, dy0 = cy - h * 0.5f;
        float dx1 = cx + w * 0.5f, dy1 = cy + h * 0.5f;
        float areab = (dx1 - dx0) * (dy1 - dy0);

        float best = -1.f; int bi = 0;
        unsigned int lowkey = 0xFFFFFFFFu - (unsigned int)p; // smaller p wins ties
        for (int m = 0; m < M_; ++m) {
            float4 t = st[m];
            float ltx = fmaxf(t.x, dx0), lty = fmaxf(t.y, dy0);
            float rbx = fminf(t.z, dx1), rby = fminf(t.w, dy1);
            float iw = fmaxf(rbx - ltx, 0.f), ih = fmaxf(rby - lty, 0.f);
            float inter = iw * ih;
            float areaa = (t.z - t.x) * (t.w - t.y);
            float iou = inter / (areaa + areab - inter);
            if (iou > best) { best = iou; bi = m; }   // first-max like jnp.argmax
            unsigned long long key =
                ((unsigned long long)__float_as_uint(iou) << 32) | lowkey;
            atomicMax(&sbm[m], key);                  // ds_max_u64
        }
        bestov [(size_t)b * P_ + p] = best;
        bestidx[(size_t)b * P_ + p] = bi;
    }
    __syncthreads();
    if (tid < M_) atomicMax(&packed[b * M_ + tid], sbm[tid]);
}

// ---------------------------------------------------------------- scatter override
__global__ void k_override(const unsigned long long* __restrict__ packed,
                           float* bestov, int* bestidx) {
    int b = blockIdx.x * blockDim.x + threadIdx.x;
    if (b >= B_) return;
    for (int m = 0; m < M_; ++m) {   // ascending: last-wins on duplicate priors
        unsigned int pi =
            0xFFFFFFFFu - (unsigned int)(packed[b * M_ + m] & 0xFFFFFFFFull);
        bestov [(size_t)b * P_ + pi] = 2.0f;
        bestidx[(size_t)b * P_ + pi] = m;
    }
}

// ---------------------------------------------------------------- fused loss pass
__global__ void k_loss(const float4* __restrict__ arm_loc,
                       const float2* __restrict__ arm_conf,
                       const float4* __restrict__ odm_loc,
                       const float*  __restrict__ odm_conf,
                       const float4* __restrict__ priors,
                       const float4* __restrict__ truths,
                       const int*    __restrict__ labels,
                       float* __restrict__ ws_ov,     // in: best_ov, out: mine
                       const int* __restrict__ bestidx,
                       int* numpos, float* sumposce, float* lossl) {
    __shared__ float   sconf[BLK * C_];
    __shared__ float4  st[M_];
    __shared__ int     slab[M_];
    __shared__ _Float16 sph[2 * BLK];
    __shared__ float   red[BLK];

    const int b   = blockIdx.y;
    const int p0  = blockIdx.x * BLK;
    const int tid = threadIdx.x;
    const int p   = p0 + tid;

    if (tid < M_) { st[tid] = truths[b * M_ + tid]; slab[tid] = labels[b * M_ + tid]; }
    // stage conf tile to LDS (coalesced); stride-21 rows are bank-conflict free
    const int rows = min(BLK, P_ - p0);
    const float* src = odm_conf + ((size_t)b * P_ + p0) * C_;
#if USE_ASYNC_LDS
    // CDNA5 async DMA path: global -> LDS directly, tracked by ASYNCcnt,
    // no VGPR staging. Each wave drains its own counter before the barrier.
    float* srcnc = const_cast<float*>(src);
    for (int i = tid; i < rows * C_; i += BLK) {
        __builtin_amdgcn_global_load_async_to_lds_b32(
            (gas1_i32p)(void*)(srcnc + i),
            (las3_i32p)(void*)(&sconf[i]), 0, 0);
    }
    wait_async0();
#else
    for (int i = tid; i < rows * C_; i += BLK) sconf[i] = src[i];
#endif
    __syncthreads();

    float sl1 = 0.f, cepos = 0.f, posf = 0.f;
    if (p < P_) {
        const size_t gi = (size_t)b * P_ + p;
        // decode (ARM-refined prior)
        float4 pr = priors[p];
        float4 l  = arm_loc[gi];
        float cx = pr.x + l.x * 0.1f * pr.z;
        float cy = pr.y + l.y * 0.1f * pr.w;
        float w  = pr.z * expf(l.z * 0.2f);
        float h  = pr.w * expf(l.w * 0.2f);
        float dx0 = cx - w * 0.5f, dy0 = cy - h * 0.5f;
        float dx1 = cx + w * 0.5f, dy1 = cy + h * 0.5f;
        // center_size round trip (match reference arithmetic)
        float ccx = (dx1 + dx0) * 0.5f, ccy = (dy1 + dy0) * 0.5f;
        float cw  = dx1 - dx0,          ch  = dy1 - dy0;

        int   ti = bestidx[gi];
        float ov = ws_ov[gi];
        float4 mt = st[ti];
        int conf_t = (ov < THRESH) ? 0 : slab[ti];

        // encode targets
        float gx = ((mt.x + mt.z) * 0.5f - ccx) / (0.1f * cw);
        float gy = ((mt.y + mt.w) * 0.5f - ccy) / (0.1f * ch);
        float gw = logf((mt.z - mt.x) / cw) / 0.2f;
        float gh = logf((mt.w - mt.y) / ch) / 0.2f;

        // ARM objectness softmax
        float2 ac = arm_conf[gi];
        float mm = fmaxf(ac.x, ac.y);
        float e0 = expf(ac.x - mm), e1 = expf(ac.y - mm);
        bool pos = (conf_t > 0) && (e1 / (e0 + e1) > THETA);
        posf = pos ? 1.f : 0.f;

        // CE = logsumexp - gold over 21 classes
        const float* row = &sconf[tid * C_];
        float mx = row[0];
        for (int c = 1; c < C_; ++c) mx = fmaxf(mx, row[c]);
        float s = 0.f;
        for (int c = 0; c < C_; ++c) s += expf(row[c] - mx);
        float ce = (mx + logf(s)) - row[conf_t];

        ws_ov[gi] = pos ? 0.f : ce;   // loss_c_mine overwrites best_ov in place
        if (pos) {
            cepos = ce;
            float4 od = odm_loc[gi];
            float d0 = fabsf(od.x - gx), d1 = fabsf(od.y - gy);
            float d2 = fabsf(od.z - gw), d3 = fabsf(od.w - gh);
            sl1 = (d0 < 1.f ? 0.5f * d0 * d0 : d0 - 0.5f)
                + (d1 < 1.f ? 0.5f * d1 * d1 : d1 - 0.5f)
                + (d2 < 1.f ? 0.5f * d2 * d2 : d2 - 0.5f)
                + (d3 < 1.f ? 0.5f * d3 * d3 : d3 - 0.5f);
        }
    }

    // ---- exact positive count on the matrix pipe (0/1 flags are exact in f16)
    sph[tid]       = (_Float16)posf;
    sph[tid + BLK] = (_Float16)0.f;
    __syncthreads();
    if (tid < 32) {                       // wave 0, EXEC all-ones as WMMA requires
        v16h a, bm;
        for (int i = 0; i < 16; ++i) { a[i] = (_Float16)1.f; bm[i] = sph[tid * 16 + i]; }
        v8f c = {};
        c = __builtin_amdgcn_wmma_f32_16x16x32_f16(false, a, false, bm,
                                                   (short)0, c, false, false);
        float t = c[0];                   // col-sum; rows identical -> 2x total over 32 lanes
        for (int off = 16; off >= 1; off >>= 1) t += __shfl_xor(t, off, 32);
        if (tid == 0) atomicAdd(&numpos[b], (int)(t * 0.5f + 0.5f));
    }

    // ---- float sums via LDS tree + f32 atomics
    red[tid] = sl1; __syncthreads();
    for (int s2 = BLK / 2; s2 > 0; s2 >>= 1) {
        if (tid < s2) red[tid] += red[tid + s2];
        __syncthreads();
    }
    if (tid == 0 && red[0] != 0.f) atomicAdd(lossl, red[0]);
    __syncthreads();
    red[tid] = cepos; __syncthreads();
    for (int s2 = BLK / 2; s2 > 0; s2 >>= 1) {
        if (tid < s2) red[tid] += red[tid + s2];
        __syncthreads();
    }
    if (tid == 0 && red[0] != 0.f) atomicAdd(&sumposce[b], red[0]);
}

// ---------------------------------------------------------------- per-batch radix select
__global__ void k_mine(const float* __restrict__ mine, const int* __restrict__ numpos,
                       const float* __restrict__ sumposce, float* lossc) {
    __shared__ unsigned int hist[256];
    __shared__ unsigned int s_sel, s_kr;
    __shared__ float redf[BLK];
    const int b = blockIdx.x;
    const int tid = threadIdx.x;
    const float* v = mine + (size_t)b * P_;

    long long k = 3LL * numpos[b];
    if (k > P_ - 1) k = P_ - 1;
    if (k <= 0) {
        if (tid == 0) atomicAdd(lossc, sumposce[b]);
        return;
    }
    unsigned int prefix = 0;
    unsigned int kr = (unsigned int)k;
    for (int pass = 0; pass < 4; ++pass) {
        const int shift = 24 - 8 * pass;
        const unsigned int himask = (pass == 0) ? 0u : (0xFFFFFFFFu << (shift + 8));
        hist[tid] = 0u;
        __syncthreads();
        for (int i = tid; i < P_; i += BLK) {
            unsigned int u = __float_as_uint(v[i]);   // ce >= 0 -> bits monotone
            if ((u & himask) == prefix)
                atomicAdd(&hist[(u >> shift) & 0xFFu], 1u);
        }
        __syncthreads();
        if (tid == 0) {
            unsigned int cum = 0, sel = 0, krn = kr;
            for (int d = 255; d >= 0; --d) {
                unsigned int c = hist[d];
                if (kr <= cum + c) { sel = (unsigned int)d; krn = kr - cum; break; }
                cum += c;
            }
            s_sel = sel; s_kr = krn;
        }
        __syncthreads();
        prefix |= (s_sel << shift);
        kr = s_kr;
        __syncthreads();
    }
    // top-k sum is tie-order invariant: sum(v > T) + kr * T
    float sumG = 0.f;
    for (int i = tid; i < P_; i += BLK) {
        unsigned int u = __float_as_uint(v[i]);
        if (u > prefix) sumG += v[i];
    }
    redf[tid] = sumG; __syncthreads();
    for (int s2 = BLK / 2; s2 > 0; s2 >>= 1) {
        if (tid < s2) redf[tid] += redf[tid + s2];
        __syncthreads();
    }
    if (tid == 0) {
        float lc = sumposce[b] + redf[0] + (float)kr * __uint_as_float(prefix);
        atomicAdd(lossc, lc);
    }
}

// ---------------------------------------------------------------- finalize
__global__ void k_final(const int* numpos, const float* lossl, const float* lossc,
                        float* out) {
    if (blockIdx.x == 0 && threadIdx.x == 0) {
        int n = 0;
        for (int b = 0; b < B_; ++b) n += numpos[b];
        float N = (float)n;
        out[0] = lossl[0] / N;
        out[1] = lossc[0] / N;
    }
}

// ---------------------------------------------------------------- launch
extern "C" void kernel_launch(void* const* d_in, const int* in_sizes, int n_in,
                              void* d_out, int out_size, void* d_ws, size_t ws_size,
                              hipStream_t stream) {
    const float4* arm_loc  = (const float4*)d_in[0];
    const float2* arm_conf = (const float2*)d_in[1];
    const float4* odm_loc  = (const float4*)d_in[2];
    const float*  odm_conf = (const float*) d_in[3];
    const float4* priors   = (const float4*)d_in[4];
    const float4* truths   = (const float4*)d_in[5];
    const int*    labels   = (const int*)   d_in[6];

    char* ws = (char*)d_ws;
    size_t off = 0;
    unsigned long long* packed = (unsigned long long*)(ws + off); off += (size_t)B_ * M_ * 8;
    float* bestov   = (float*)(ws + off); off += (size_t)B_ * P_ * 4;   // -> mine
    int*   bestidx  = (int*)  (ws + off); off += (size_t)B_ * P_ * 4;
    int*   numpos   = (int*)  (ws + off); off += (size_t)B_ * 4;
    float* sumposce = (float*)(ws + off); off += (size_t)B_ * 4;
    float* lossl    = (float*)(ws + off); off += 4;
    float* lossc    = (float*)(ws + off); off += 4;
    float* out      = (float*)d_out;

    k_init<<<(B_ * M_ + 255) / 256, 256, 0, stream>>>(packed, numpos, sumposce, lossl, lossc);
    k_match<<<dim3(PBLK, B_), BLK, 0, stream>>>(arm_loc, priors, truths,
                                                bestov, bestidx, packed);
    k_override<<<2, 32, 0, stream>>>(packed, bestov, bestidx);
    k_loss<<<dim3(PBLK, B_), BLK, 0, stream>>>(arm_loc, arm_conf, odm_loc, odm_conf,
                                               priors, truths, labels,
                                               bestov, bestidx,
                                               numpos, sumposce, lossl);
    k_mine<<<B_, BLK, 0, stream>>>(bestov, numpos, sumposce, lossc);
    k_final<<<1, 32, 0, stream>>>(numpos, lossl, lossc, out);
}
